// HybridLayer_65481071407654
// MI455X (gfx1250) — matrix operand order repaired and checked by reading, not compiled, not run on hardware
//
#include <hip/hip_runtime.h>
#include <hip/hip_bf16.h>
#include <stdint.h>

#define T_DIM 1024
#define B_DIM 128
#define D_DIM 512
#define H_DIM 256
#define TB_DIM (T_DIM * B_DIM)

typedef __attribute__((ext_vector_type(16))) __bf16    v16bf;
typedef __attribute__((ext_vector_type(8)))  float     v8f;
typedef __attribute__((ext_vector_type(4)))  unsigned  u32x4;
typedef __attribute__((ext_vector_type(8)))  unsigned  u32x8;

union FragU { v16bf v; uint4 q[2]; };

__device__ __forceinline__ unsigned short f2bf(float f) {
  unsigned u = __float_as_uint(f);
  u += 0x7FFFu + ((u >> 16) & 1u);          // round-to-nearest-even
  return (unsigned short)(u >> 16);
}
__device__ __forceinline__ float bf2f(unsigned short h) {
  return __uint_as_float(((unsigned)h) << 16);
}
__device__ __forceinline__ float sigmoidf(float x) {
  return 1.0f / (1.0f + __expf(-x));
}

__device__ __forceinline__ v8f wmma_bf16(v16bf a, v16bf b, v8f c) {
  return __builtin_amdgcn_wmma_f32_16x16x32_bf16(false, a, false, b, (short)0, c,
                                                 false, false);
}

// ---------------------------------------------------------------------------
// Tensor Data Mover: async 2-D tile load Global->LDS (ISA 08_async_tensor §8).
// D# group0 (4 SGPRs) + group1 (8 SGPRs); VADDR2/3 = NULL -> 2-D tensor.
// data_size_enc: 0=1B 1=2B 2=4B 3=8B. Dims/strides in data_size units.
// Tracked with TENSORcnt; EXEC ignored (issue from one wave).
// ---------------------------------------------------------------------------
__device__ __forceinline__ void tdm_load_2d(unsigned lds_byte_addr, const void* gaddr,
                                            unsigned tile_w, unsigned tile_h,
                                            unsigned tensor_w, unsigned tensor_h,
                                            unsigned row_stride, unsigned data_size_enc)
{
  unsigned long long ga = (unsigned long long)(uintptr_t)gaddr;
  u32x4 g0;
  g0.x = 1u;                                              // count=1 (valid, user mode)
  g0.y = lds_byte_addr;                                   // lds_addr [63:32]
  g0.z = (unsigned)ga;                                    // global_addr lo
  g0.w = ((unsigned)(ga >> 32) & 0x01FFFFFFu) | (2u << 30); // global_addr hi | type=2
  u32x8 g1;
  g1.s0 = (data_size_enc & 3u) << 16;                     // wg_mask=0, data_size
  g1.s1 = (tensor_w & 0xFFFFu) << 16;                     // tensor_dim0[15:0] @ bits 63:48
  g1.s2 = (tensor_w >> 16) | ((tensor_h & 0xFFFFu) << 16);// dim0[31:16] | dim1[15:0]
  g1.s3 = (tensor_h >> 16) | ((tile_w & 0xFFFFu) << 16);  // dim1[31:16] | tile_dim0
  g1.s4 = tile_h & 0xFFFFu;                               // tile_dim1 (tile_dim2=0)
  g1.s5 = row_stride;                                     // tensor_dim0_stride[31:0]
  g1.s6 = 0u;                                             // stride[47:32] | dim1_stride lo
  g1.s7 = 0u;
  asm volatile("tensor_load_to_lds %0, %1" :: "s"(g0), "s"(g1) : "memory");
}

// A fragment: 16x32 bf16 tile, row-major, leading dim `ld` (elements).
// Lanes 0-15: K = 0..7 & 16..23 ; lanes 16-31: K = 8..15 & 24..31 (ISA 7.12.2).
__device__ __forceinline__ v16bf load_a_frag(const unsigned short* A, int ld, int lane) {
  int m  = lane & 15;
  int kh = (lane >> 4) << 3;
  const unsigned short* p = A + m * ld + kh;
  FragU f;
  f.q[0] = *(const uint4*)(p);
  f.q[1] = *(const uint4*)(p + 16);
  return f.v;
}

// B fragment: 32x16 (KxN); source is W[N,K] row-major, i.e. B(k,n) = W[n,k].
__device__ __forceinline__ v16bf load_b_frag(const unsigned short* W, int ld, int lane) {
  int n  = lane & 15;
  int kh = (lane >> 4) << 3;
  const unsigned short* p = W + n * ld + kh;
  FragU f;
  f.q[0] = *(const uint4*)(p);
  f.q[1] = *(const uint4*)(p + 16);
  return f.v;
}

// ---------------------------------------------------------------------------
__global__ void cvt_f32_bf16(const float* __restrict__ s, unsigned short* __restrict__ d, int n) {
  for (int i = blockIdx.x * blockDim.x + threadIdx.x; i < n; i += gridDim.x * blockDim.x)
    d[i] = f2bf(s[i]);
}

// ---------------------------------------------------------------------------
// Tiled WMMA GEMM:  Cout[M,N](bf16) = act(A[M,K] @ W[N,K]^T + bias[N])
// ACT: 0 = tanh, 1 = relu.  Tiles staged via TDM (double-buffered); fp32 A is
// converted to bf16 manually while B rides the TDM.
// Block = 256 threads (8 waves), tile = 128x64, K step = 32.
// ---------------------------------------------------------------------------
template<int ACT, bool A_IS_F32>
__global__ __launch_bounds__(256)
void gemm_bias_act(const void* __restrict__ Av, const unsigned short* __restrict__ W,
                   const float* __restrict__ bias, unsigned short* __restrict__ Cout,
                   int M, int N, int K)
{
  __shared__ unsigned short Alds[2][128][32];   // 16 KB (double-buffered)
  __shared__ unsigned short Blds[2][64][32];    //  8 KB

  const int tid  = threadIdx.x;
  const int lane = tid & 31;
  const int wave = tid >> 5;
  const int m0 = blockIdx.y * 128;
  const int n0 = blockIdx.x * 64;
  const int nk = K >> 5;

  v8f c[4];
  #pragma unroll
  for (int j = 0; j < 4; ++j)
    #pragma unroll
    for (int r = 0; r < 8; ++r) c[j][r] = 0.0f;

  const int arow = tid >> 1;
  const int acb  = (tid & 1) << 4;

  // ---- prologue: issue TDM tile(0) ----
  if (wave == 0) {
    tdm_load_2d((unsigned)(uintptr_t)&Blds[0][0][0], W + (size_t)n0 * K,
                32, 64, (unsigned)K, (unsigned)(N - n0), (unsigned)K, 1u);
    if (!A_IS_F32)
      tdm_load_2d((unsigned)(uintptr_t)&Alds[0][0][0],
                  (const unsigned short*)Av + (size_t)m0 * K,
                  32, 128, (unsigned)K, (unsigned)(M - m0), (unsigned)K, 1u);
  }

  for (int kidx = 0; kidx < nk; ++kidx) {
    const int k   = kidx << 5;
    const int buf = kidx & 1;

    // fp32 A: convert + stage manually into the current buffer
    if (A_IS_F32) {
      const float* Ap = (const float*)Av + (size_t)(m0 + arow) * K + k + acb;
      __builtin_prefetch(Ap + 32, 0, 0);
      #pragma unroll
      for (int i = 0; i < 4; ++i) {
        float4 f = *(const float4*)(Ap + 4 * i);
        uint2 o;
        o.x = (unsigned)f2bf(f.x) | ((unsigned)f2bf(f.y) << 16);
        o.y = (unsigned)f2bf(f.z) | ((unsigned)f2bf(f.w) << 16);
        *(uint2*)&Alds[buf][arow][acb + 4 * i] = o;
      }
    }

    if (wave == 0) __builtin_amdgcn_s_wait_tensorcnt(0);  // tile(kidx) DMA done
    __syncthreads();                                      // LDS visible to all waves

    // overlap: issue DMA for tile(kidx+1) into the other buffer
    if (wave == 0 && (kidx + 1) < nk) {
      const int kn = k + 32;
      tdm_load_2d((unsigned)(uintptr_t)&Blds[buf ^ 1][0][0], W + (size_t)n0 * K + kn,
                  32, 64, (unsigned)(K - kn), (unsigned)(N - n0), (unsigned)K, 1u);
      if (!A_IS_F32)
        tdm_load_2d((unsigned)(uintptr_t)&Alds[buf ^ 1][0][0],
                    (const unsigned short*)Av + (size_t)m0 * K + kn,
                    32, 128, (unsigned)(K - kn), (unsigned)(M - m0), (unsigned)K, 1u);
    }

    // ---- WMMA on current buffer ----
    v16bf a = load_a_frag(&Alds[buf][wave * 16][0], 32, lane);
    #pragma unroll
    for (int j = 0; j < 4; ++j) {
      v16bf b = load_b_frag(&Blds[buf][j * 16][0], 32, lane);
      c[j] = wmma_bf16(a, b, c[j]);
    }
  }

  // ---- epilogue: bias + activation + bf16 store ----
  const int mrow = m0 + wave * 16 + ((lane >> 4) << 3);
  #pragma unroll
  for (int j = 0; j < 4; ++j) {
    const int n = n0 + j * 16 + (lane & 15);
    const float bv = bias[n];
    #pragma unroll
    for (int r = 0; r < 8; ++r) {
      float x = c[j][r] + bv;
      x = (ACT == 0) ? tanhf(x) : fmaxf(x, 0.0f);
      Cout[(size_t)(mrow + r) * N + n] = f2bf(x);
    }
  }
}

// ---------------------------------------------------------------------------
// Persistent LSTM scan. 16 workgroups (one WGP each / one cluster's worth).
// Block j owns h-columns [16j,16j+16) for ALL 4 gates, so c/h updates are
// WGP-local. Per step: [128,512] @ [512,64]^T via WMMA (K = [h_t | h_{t-1}]),
// then gate math, write h_t slice, cluster barrier + atomic grid barrier.
// ---------------------------------------------------------------------------
__global__ __launch_bounds__(256)
void lstm_scan(const unsigned short* __restrict__ h,
               const unsigned short* __restrict__ Wf, const unsigned short* __restrict__ Wi,
               const unsigned short* __restrict__ Wu, const unsigned short* __restrict__ Wo,
               const float* __restrict__ bf_, const float* __restrict__ bi_,
               const float* __restrict__ bu_, const float* __restrict__ bo_,
               unsigned short* __restrict__ hs, unsigned int* __restrict__ sync_cnt)
{
  __shared__ float c_lds[B_DIM][17];

  const int tid  = threadIdx.x;
  const int lane = tid & 31;
  const int wave = tid >> 5;
  const int ncol0 = blockIdx.x * 16;

  for (int i = tid; i < B_DIM * 16; i += 256) c_lds[i >> 4][i & 15] = 0.0f;

  const int nl = lane & 15;
  const float bfv = bf_[ncol0 + nl], biv = bi_[ncol0 + nl];
  const float buv = bu_[ncol0 + nl], bov = bo_[ncol0 + nl];
  const int mbase = wave * 16;
  __syncthreads();

  for (int t = 0; t < T_DIM; ++t) {
    v8f cf, ci, cu, co;
    #pragma unroll
    for (int r = 0; r < 8; ++r) { cf[r] = bfv; ci[r] = biv; cu[r] = buv; co[r] = bov; }

    const unsigned short* hx = h + (size_t)t * B_DIM * H_DIM;                       // K 0..255
    const unsigned short* hp = (t > 0) ? hs + (size_t)(t - 1) * B_DIM * H_DIM : hx; // K 256..511
    const int kmax = (t == 0) ? 8 : 16;     // h_{-1} == 0: skip its K half

    for (int kk = 0; kk < kmax; ++kk) {
      const int kb = kk * 32;
      v16bf a = (kk < 8)
        ? load_a_frag(hx + mbase * H_DIM + kb,         H_DIM, lane)
        : load_a_frag(hp + mbase * H_DIM + (kb - 256), H_DIM, lane);
      v16bf b;
      b = load_b_frag(Wf + (size_t)ncol0 * 512 + kb, 512, lane); cf = wmma_bf16(a, b, cf);
      b = load_b_frag(Wi + (size_t)ncol0 * 512 + kb, 512, lane); ci = wmma_bf16(a, b, ci);
      b = load_b_frag(Wu + (size_t)ncol0 * 512 + kb, 512, lane); cu = wmma_bf16(a, b, cu);
      b = load_b_frag(Wo + (size_t)ncol0 * 512 + kb, 512, lane); co = wmma_bf16(a, b, co);
    }

    unsigned short* hout = hs + (size_t)t * B_DIM * H_DIM;
    const int mhi = mbase + ((lane >> 4) << 3);
    #pragma unroll
    for (int r = 0; r < 8; ++r) {
      const int m = mhi + r;
      float f = sigmoidf(cf[r]);
      float i = sigmoidf(ci[r]);
      float g = tanhf(cu[r]);
      float o = sigmoidf(co[r]);
      float cn = f * c_lds[m][nl] + i * g;
      c_lds[m][nl] = cn;
      hout[(size_t)m * H_DIM + ncol0 + nl] = f2bf(o * tanhf(cn));
    }

    // make h(t) visible to all 16 WGPs before step t+1
    __threadfence();
    __syncthreads();
    __builtin_amdgcn_s_cluster_barrier();    // cluster split-barrier (NOP if not a cluster)
    if (tid == 0) {
      __hip_atomic_fetch_add(sync_cnt, 1u, __ATOMIC_ACQ_REL, __HIP_MEMORY_SCOPE_AGENT);
      const unsigned target = 16u * (unsigned)(t + 1);
      while (__hip_atomic_load(sync_cnt, __ATOMIC_ACQUIRE, __HIP_MEMORY_SCOPE_AGENT) < target)
        __builtin_amdgcn_s_sleep(1);
    }
    __syncthreads();
  }
}

// ---------------------------------------------------------------------------
__global__ __launch_bounds__(256)
void head_gemv(const unsigned short* __restrict__ z, const float* __restrict__ w2,
               const float* __restrict__ b2, float* __restrict__ out, int M)
{
  const int idx = blockIdx.x * 256 + threadIdx.x;
  if (idx >= M * 2) return;
  const int m = idx >> 1, j = idx & 1;
  const unsigned short* zp = z + (size_t)m * H_DIM;
  const float* wp = w2 + j * H_DIM;
  float acc = b2[j];
  #pragma unroll 8
  for (int k = 0; k < H_DIM; ++k) acc += bf2f(zp[k]) * wp[k];
  out[idx] = acc;
}

// ---------------------------------------------------------------------------
extern "C" void kernel_launch(void* const* d_in, const int* in_sizes, int n_in,
                              void* d_out, int out_size, void* d_ws, size_t ws_size,
                              hipStream_t stream)
{
  const float* x        = (const float*)d_in[0];
  const float* fc_w     = (const float*)d_in[1];
  const float* fc_b     = (const float*)d_in[2];
  const float* forget_w = (const float*)d_in[3];
  const float* forget_b = (const float*)d_in[4];
  const float* input_w  = (const float*)d_in[5];
  const float* input_b  = (const float*)d_in[6];
  const float* update_w = (const float*)d_in[7];
  const float* update_b = (const float*)d_in[8];
  const float* output_w = (const float*)d_in[9];
  const float* output_b = (const float*)d_in[10];
  const float* cls_w0   = (const float*)d_in[11];
  const float* cls_b0   = (const float*)d_in[12];
  const float* cls_w1   = (const float*)d_in[13];
  const float* cls_b1   = (const float*)d_in[14];
  const float* cls_w2   = (const float*)d_in[15];
  const float* cls_b2   = (const float*)d_in[16];

  unsigned short* wFC = (unsigned short*)d_ws;            // H*D      = 131072
  unsigned short* wF  = wFC + 131072;                     // H*2H     = 131072
  unsigned short* wI  = wF  + 131072;
  unsigned short* wU  = wI  + 131072;
  unsigned short* wO  = wU  + 131072;
  unsigned short* wC0 = wO  + 131072;                     // H*H      = 65536
  unsigned short* wC1 = wC0 + 65536;
  unsigned short* hbuf  = wC1 + 65536;                    // [T*B, H] bf16 (reused for z1)
  unsigned short* hsbuf = hbuf + (size_t)TB_DIM * H_DIM;  // [T*B, H] bf16 (reused for z2)
  unsigned int*   cnt   = (unsigned int*)(hsbuf + (size_t)TB_DIM * H_DIM);

  hipMemsetAsync(cnt, 0, 64, stream);

  cvt_f32_bf16<<<256, 256, 0, stream>>>(fc_w,     wFC, 131072);
  cvt_f32_bf16<<<256, 256, 0, stream>>>(forget_w, wF,  131072);
  cvt_f32_bf16<<<256, 256, 0, stream>>>(input_w,  wI,  131072);
  cvt_f32_bf16<<<256, 256, 0, stream>>>(update_w, wU,  131072);
  cvt_f32_bf16<<<256, 256, 0, stream>>>(output_w, wO,  131072);
  cvt_f32_bf16<<<256, 256, 0, stream>>>(cls_w0,   wC0, 65536);
  cvt_f32_bf16<<<256, 256, 0, stream>>>(cls_w1,   wC1, 65536);

  // h = tanh(x @ fc_w^T + fc_b)
  gemm_bias_act<0, true ><<<dim3(H_DIM / 64, TB_DIM / 128), 256, 0, stream>>>(
      x, wFC, fc_b, hbuf, TB_DIM, H_DIM, D_DIM);

  // sequential gated recurrence, 16 WGPs, WMMA per step
  lstm_scan<<<16, 256, 0, stream>>>(hbuf, wF, wI, wU, wO,
                                    forget_b, input_b, update_b, output_b,
                                    hsbuf, cnt);

  // z1 = relu(hs @ w0^T + b0)
  gemm_bias_act<1, false><<<dim3(H_DIM / 64, TB_DIM / 128), 256, 0, stream>>>(
      hsbuf, wC0, cls_b0, hbuf, TB_DIM, H_DIM, H_DIM);
  // z2 = relu(z1 @ w1^T + b1)
  gemm_bias_act<1, false><<<dim3(H_DIM / 64, TB_DIM / 128), 256, 0, stream>>>(
      hbuf, wC1, cls_b1, hsbuf, TB_DIM, H_DIM, H_DIM);

  // logits = z2 @ w2^T + b2  -> fp32 [T,B,2]
  head_gemv<<<(TB_DIM * 2 + 255) / 256, 256, 0, stream>>>(
      hsbuf, cls_w2, cls_b2, (float*)d_out, TB_DIM);
}
